// TorchBoostModel_81784767250546
// MI455X (gfx1250) — matrix operand order repaired and checked by reading, not compiled, not run on hardware
//
#include <hip/hip_runtime.h>
#include <hip/hip_bf16.h>

// ---------------------------------------------------------------------------
// TorchBoost forward for MI455X (gfx1250, wave32, WMMA + async-LDS).
//   B=16384, D=512, T=64 trees, depth 6 (63 nodes / 64 leaves), 10 classes,
//   attention MLP 512->128->64.
// Big GEMM (x @ tree_w^T, 67.7 GFLOP) via v_wmma_f32_16x16x32_bf16 with
// bf16 hi/lo split (3 wmma per tile) for ~fp32 accuracy.
// Leaf reduction via v_wmma_f32_16x16x4_f32 (exact fp32).
// leaf_values / attn tiles staged with global_load_async_to_lds_b128
// (ASYNCcnt), overlapped with the WMMA GEMM.
// ---------------------------------------------------------------------------

typedef __bf16 bf16_t;
typedef __attribute__((ext_vector_type(16))) __bf16 v16bf;
typedef __attribute__((ext_vector_type(8)))  __bf16 v8bf;
typedef __attribute__((ext_vector_type(8)))  float  v8f;
typedef __attribute__((ext_vector_type(2)))  float  v2f;

#define NBATCH   16384
#define DIN      512
#define NTREES   64
#define NNODES   63
#define NLEAF    64
#define NCLS     10
#define NHID     128

__device__ __forceinline__ void splitbf(float v, bf16_t& h, bf16_t& l) {
    h = (bf16_t)v;
    l = (bf16_t)(v - (float)h);
}

// A operand (16x32 bf16): lane holds K chunks [p..p+7] and [p+16..p+23]
__device__ __forceinline__ v16bf loadA16(const bf16_t* p) {
    v8bf a = *(const v8bf*)p;
    v8bf b = *(const v8bf*)(p + 16);
    v16bf r;
#pragma unroll
    for (int i = 0; i < 8; ++i) { r[i] = a[i]; r[i + 8] = b[i]; }
    return r;
}
// B operand (32x16 bf16): lane holds K chunks [p..p+7] and [p+16..p+23]
__device__ __forceinline__ v16bf loadB16(const bf16_t* p) {
    v8bf a = *(const v8bf*)p;
    v8bf b = *(const v8bf*)(p + 16);
    v16bf r;
#pragma unroll
    for (int i = 0; i < 8; ++i) { r[i] = a[i]; r[i + 8] = b[i]; }
    return r;
}

__device__ __forceinline__ v8f wmma_bf16(v16bf a, v16bf b, v8f c) {
    return __builtin_amdgcn_wmma_f32_16x16x32_bf16(false, a, false, b, (short)0, c,
                                                   false, false);
}
__device__ __forceinline__ v8f wmma_f32x4(v2f a, v2f b, v8f c) {
    return __builtin_amdgcn_wmma_f32_16x16x4_f32(false, a, false, b, (short)0, c,
                                                 false, false);
}
__device__ __forceinline__ void wavebar() {
    __asm__ volatile("" ::: "memory");
    __builtin_amdgcn_wave_barrier();
    __asm__ volatile("" ::: "memory");
}
__device__ __forceinline__ float fsigmoid(float z) {
    return 1.0f / (1.0f + __expf(-z));
}

// Async global -> LDS copy, 16B per lane (gfx1250 ASYNCcnt path).
// Generic pointers to LDS carry the LDS byte offset in their low 32 bits.
__device__ __forceinline__ void async_ld_b128(void* lds_dst, const void* gsrc) {
    unsigned ldsa = (unsigned)(uintptr_t)lds_dst;
    asm volatile("global_load_async_to_lds_b128 %0, %1, off"
                 :: "v"(ldsa), "v"(gsrc) : "memory");
}
__device__ __forceinline__ void wait_async0() {
    asm volatile("s_wait_asynccnt 0x0" ::: "memory");
}

// ---------------------------------------------------------------------------
// pack_x: one wave per row. NaN-clean, bf16 hi/lo split, per-row missing flag.
__global__ __launch_bounds__(128) void pack_x_kernel(const float* __restrict__ x,
                                                     bf16_t* __restrict__ xhi,
                                                     bf16_t* __restrict__ xlo,
                                                     unsigned* __restrict__ missing) {
    int gwave = (blockIdx.x * 128 + threadIdx.x) >> 5;   // == row
    int lane  = threadIdx.x & 31;
    const float* xr = x + (size_t)gwave * DIN;
    bool any = false;
#pragma unroll 4
    for (int j = 0; j < DIN / 32; ++j) {
        int e = j * 32 + lane;
        float v = xr[e];
        bool nan = (v != v);
        any |= nan;
        if (nan) v = 0.0f;
        bf16_t h, l;
        splitbf(v, h, l);
        xhi[(size_t)gwave * DIN + e] = h;
        xlo[(size_t)gwave * DIN + e] = l;
    }
    bool a = __any((int)any);
    if (lane == 0) missing[gwave] = a ? 1u : 0u;
}

// pack_tw: tree_w [64][63][512] -> bf16 hi/lo [64][64][512] (node 63 zero pad)
__global__ __launch_bounds__(128) void pack_tw_kernel(const float* __restrict__ tw,
                                                      bf16_t* __restrict__ twhi,
                                                      bf16_t* __restrict__ twlo) {
    int idx = blockIdx.x * 128 + threadIdx.x;          // 64*64*512 total
    int d = idx & (DIN - 1);
    int node = (idx >> 9) & 63;
    int t = idx >> 15;
    float v = (node < NNODES) ? tw[((size_t)t * NNODES + node) * DIN + d] : 0.0f;
    bf16_t h, l;
    splitbf(v, h, l);
    twhi[idx] = h;
    twlo[idx] = l;
}

// pack_lv: leaf_values [64][64][10] -> padded f32 [64][64][16] (contiguous
// 4KB per tree so the main kernel can DMA it to LDS with async b128 copies)
__global__ __launch_bounds__(128) void pack_lv_kernel(const float* __restrict__ lv,
                                                      float* __restrict__ lvpad) {
    int idx = blockIdx.x * 128 + threadIdx.x;          // 64*64*16 total
    int c = idx & 15;
    int leaf = (idx >> 4) & 63;
    int t = idx >> 10;
    lvpad[idx] =
        (c < NCLS) ? lv[((size_t)t * NLEAF + leaf) * NCLS + c] : 0.0f;
}

// pack_small: w1^T, w2^T to bf16 hi/lo, fold BN into scale/shift
__global__ __launch_bounds__(128) void pack_small_kernel(
    const float* __restrict__ w1, const float* __restrict__ w2,
    const float* __restrict__ gamma, const float* __restrict__ beta,
    const float* __restrict__ mean, const float* __restrict__ var,
    bf16_t* __restrict__ w1thi, bf16_t* __restrict__ w1tlo,
    bf16_t* __restrict__ w2thi, bf16_t* __restrict__ w2tlo,
    float* __restrict__ bns, float* __restrict__ bnt) {
    int idx = blockIdx.x * 128 + threadIdx.x;
    if (idx < DIN * NHID) {                     // w1t[j][d] = w1[d][j]
        int j = idx >> 9, d = idx & (DIN - 1);
        bf16_t h, l;
        splitbf(w1[d * NHID + j], h, l);
        w1thi[idx] = h; w1tlo[idx] = l;
    } else if (idx < DIN * NHID + NTREES * NHID) {   // w2t[t][j] = w2[j][t]
        int i = idx - DIN * NHID;
        int t = i >> 7, j = i & (NHID - 1);
        bf16_t h, l;
        splitbf(w2[j * NTREES + t], h, l);
        w2thi[i] = h; w2tlo[i] = l;
    } else {
        int j = idx - (DIN * NHID + NTREES * NHID);
        if (j < NHID) {
            float s = gamma[j] * rsqrtf(var[j] + 1e-5f);
            bns[j] = s;
            bnt[j] = beta[j] - mean[j] * s;
        }
    }
}

// ---------------------------------------------------------------------------
// Attention: per wave one 16-row tile. relu(x@W1+b1) -> BN -> @W2+b2 -> softmax
#define HSTRIDE 136   // 128 + 8 pad (bank-conflict-free b128 LDS loads)
__global__ __launch_bounds__(128) void attn_kernel(
    const bf16_t* __restrict__ xhi, const bf16_t* __restrict__ xlo,
    const bf16_t* __restrict__ w1thi, const bf16_t* __restrict__ w1tlo,
    const bf16_t* __restrict__ w2thi, const bf16_t* __restrict__ w2tlo,
    const float* __restrict__ b1, const float* __restrict__ bns,
    const float* __restrict__ bnt, const float* __restrict__ b2,
    float* __restrict__ attn) {
    __shared__ bf16_t hhi[4][16 * HSTRIDE];
    __shared__ bf16_t hlo[4][16 * HSTRIDE];
    int tid = threadIdx.x, wave = tid >> 5, lane = tid & 31;
    int l16 = lane & 15, half = lane >> 4;
    int m = blockIdx.x * 4 + wave;
    int row0 = m * 16;
    bf16_t* Hh = hhi[wave];
    bf16_t* Hl = hlo[wave];

    // ---- GEMM1: [16x512] @ [512x128]
    v8f acc[8];
#pragma unroll
    for (int nt = 0; nt < 8; ++nt) acc[nt] = (v8f){};
    const bf16_t* ah = xhi + (size_t)(row0 + l16) * DIN + half * 8;
    const bf16_t* al = xlo + (size_t)(row0 + l16) * DIN + half * 8;
    for (int ks = 0; ks < DIN / 32; ++ks) {
        v16bf Ahi = loadA16(ah + ks * 32);
        v16bf Alo = loadA16(al + ks * 32);
#pragma unroll
        for (int nt = 0; nt < 8; ++nt) {
            size_t boff = (size_t)(nt * 16 + l16) * DIN + ks * 32 + half * 8;
            v16bf Bhi = loadB16(w1thi + boff);
            v16bf Blo = loadB16(w1tlo + boff);
            acc[nt] = wmma_bf16(Ahi, Bhi, acc[nt]);
            acc[nt] = wmma_bf16(Ahi, Blo, acc[nt]);
            acc[nt] = wmma_bf16(Alo, Bhi, acc[nt]);
        }
    }
    // epilogue: +b1, relu, BN, split -> LDS
#pragma unroll
    for (int nt = 0; nt < 8; ++nt) {
        int col = nt * 16 + l16;
        float bias = b1[col], s = bns[col], tshift = bnt[col];
#pragma unroll
        for (int r = 0; r < 8; ++r) {
            float v = acc[nt][r] + bias;
            v = fmaxf(v, 0.0f);
            v = v * s + tshift;
            bf16_t hv, lv;
            splitbf(v, hv, lv);
            int row16 = r + 8 * half;
            Hh[row16 * HSTRIDE + col] = hv;
            Hl[row16 * HSTRIDE + col] = lv;
        }
    }
    wavebar();

    // ---- GEMM2: [16x128] @ [128x64]
    v8f a2[4];
#pragma unroll
    for (int nt = 0; nt < 4; ++nt) a2[nt] = (v8f){};
#pragma unroll
    for (int ks = 0; ks < NHID / 32; ++ks) {
        v16bf Ahi = loadA16(Hh + l16 * HSTRIDE + ks * 32 + half * 8);
        v16bf Alo = loadA16(Hl + l16 * HSTRIDE + ks * 32 + half * 8);
#pragma unroll
        for (int nt = 0; nt < 4; ++nt) {
            size_t boff = (size_t)(nt * 16 + l16) * NHID + ks * 32 + half * 8;
            v16bf Bhi = loadB16(w2thi + boff);
            v16bf Blo = loadB16(w2tlo + boff);
            a2[nt] = wmma_bf16(Ahi, Bhi, a2[nt]);
            a2[nt] = wmma_bf16(Ahi, Blo, a2[nt]);
            a2[nt] = wmma_bf16(Alo, Bhi, a2[nt]);
        }
    }
    // +b2, softmax over 64 columns per row (rows live in 16-lane halves)
#pragma unroll
    for (int nt = 0; nt < 4; ++nt) {
        float bias = b2[nt * 16 + l16];
#pragma unroll
        for (int r = 0; r < 8; ++r) a2[nt][r] += bias;
    }
    float mr[8], sr[8];
#pragma unroll
    for (int r = 0; r < 8; ++r)
        mr[r] = fmaxf(fmaxf(a2[0][r], a2[1][r]), fmaxf(a2[2][r], a2[3][r]));
#pragma unroll
    for (int d = 1; d < 16; d <<= 1)
#pragma unroll
        for (int r = 0; r < 8; ++r)
            mr[r] = fmaxf(mr[r], __shfl_xor(mr[r], d, 32));
#pragma unroll
    for (int r = 0; r < 8; ++r) sr[r] = 0.0f;
#pragma unroll
    for (int nt = 0; nt < 4; ++nt)
#pragma unroll
        for (int r = 0; r < 8; ++r) {
            float e = __expf(a2[nt][r] - mr[r]);
            a2[nt][r] = e;
            sr[r] += e;
        }
#pragma unroll
    for (int d = 1; d < 16; d <<= 1)
#pragma unroll
        for (int r = 0; r < 8; ++r) sr[r] += __shfl_xor(sr[r], d, 32);
#pragma unroll
    for (int r = 0; r < 8; ++r) sr[r] = 1.0f / sr[r];
#pragma unroll
    for (int nt = 0; nt < 4; ++nt)
#pragma unroll
        for (int r = 0; r < 8; ++r) {
            int row = row0 + r + 8 * half;
            attn[(size_t)row * NTREES + nt * 16 + l16] = a2[nt][r] * sr[r];
        }
}

// ---------------------------------------------------------------------------
// Main: per wave (16-row tile, 16 trees). Logits via bf16 WMMA, sigmoid,
// routing, leaf reduction via f32 WMMA, attn-weighted accumulate, softmax.
// leaf_values / attn tiles arrive via async global->LDS copies.
#define DSTRIDE 65
__global__ __launch_bounds__(128) void tree_main_kernel(
    const bf16_t* __restrict__ xhi, const bf16_t* __restrict__ xlo,
    const bf16_t* __restrict__ twhi, const bf16_t* __restrict__ twlo,
    const float* __restrict__ attn, const unsigned* __restrict__ missing,
    const float* __restrict__ tree_b, const float* __restrict__ tree_temp,
    const float* __restrict__ resw, const float* __restrict__ lvpad,
    float* __restrict__ out) {
    __shared__ float attn_s[16 * 64];
    __shared__ float pred_s[16 * 16];
    __shared__ float dec_s[4][16 * DSTRIDE];
    __shared__ float rout_s[4][16 * DSTRIDE];
    __shared__ float lv_s[4][64 * 16];

    int tid = threadIdx.x, wave = tid >> 5, lane = tid & 31;
    int l16 = lane & 15, half = lane >> 4;
    int m = blockIdx.x;
    int row0 = m * 16;

    // attn tile (contiguous 4KB in global) via async DMA; each wave waits its
    // own ASYNCcnt before the block barrier.
    {
        const char* ag = (const char*)(attn + (size_t)row0 * NTREES);
        char* as = (char*)attn_s;
#pragma unroll
        for (int j = 0; j < 2; ++j) {
            int cidx = tid + j * 128;       // 256 x 16B chunks
            async_ld_b128(as + cidx * 16, ag + cidx * 16);
        }
    }
    for (int i = tid; i < 256; i += 128) pred_s[i] = 0.0f;
    wait_async0();
    __syncthreads();

    unsigned mmask =
        (unsigned)(__ballot((lane < 16) && (missing[row0 + l16] != 0u)) & 0xFFFFu);

    float* decw = dec_s[wave];
    float* routw = rout_s[wave];
    float* lvw = lv_s[wave];

    const bf16_t* arow_hi = xhi + (size_t)(row0 + l16) * DIN + half * 8;
    const bf16_t* arow_lo = xlo + (size_t)(row0 + l16) * DIN + half * 8;

    for (int tt = 0; tt < 16; ++tt) {
        int t = wave * 16 + tt;
        float invtemp = 1.0f / tree_temp[t];
        float coef = 0.3f * resw[t];

        // kick off leaf-value tile DMA (16KB, contiguous); hides under GEMM
        {
            const char* lg = (const char*)(lvpad + (size_t)t * NLEAF * 16);
            char* ll = (char*)lvw;
#pragma unroll
            for (int j = 0; j < 32; ++j) {
                int cidx = lane + j * 32;   // 1024 x 16B chunks per tree
                async_ld_b128(ll + cidx * 16, lg + cidx * 16);
            }
        }

        // ---- GEMM: logits [16 rows x 64 nodes]
        v8f acc[4];
#pragma unroll
        for (int nt = 0; nt < 4; ++nt) acc[nt] = (v8f){};
        for (int ks = 0; ks < DIN / 32; ++ks) {
            v16bf Ahi = loadA16(arow_hi + ks * 32);
            v16bf Alo = loadA16(arow_lo + ks * 32);
#pragma unroll
            for (int nt = 0; nt < 4; ++nt) {
                size_t boff = ((size_t)t * 64 + nt * 16 + l16) * DIN + ks * 32 +
                              half * 8;
                v16bf Bhi = loadB16(twhi + boff);
                v16bf Blo = loadB16(twlo + boff);
                acc[nt] = wmma_bf16(Ahi, Bhi, acc[nt]);
                acc[nt] = wmma_bf16(Ahi, Blo, acc[nt]);
                acc[nt] = wmma_bf16(Alo, Bhi, acc[nt]);
            }
        }

        // ---- sigmoid decisions -> LDS
#pragma unroll
        for (int nt = 0; nt < 4; ++nt) {
            int col = nt * 16 + l16;
            float bias = (col < NNODES) ? tree_b[t * NNODES + col] : 0.0f;
#pragma unroll
            for (int r = 0; r < 8; ++r) {
                float z = (acc[nt][r] + bias) * invtemp;
                float d = fsigmoid(z);
                int row16 = r + 8 * half;
                if ((mmask >> row16) & 1u) d = 0.5f;
                decw[row16 * DSTRIDE + col] = d;
            }
        }
        wavebar();

        // ---- routing: leaves (lane, lane+32) share levels 0..4
        for (int r = 0; r < 16; ++r) {
            float p = 1.0f;
            int i = lane;
#pragma unroll
            for (int lvl = 0; lvl < 5; ++lvl) {
                int node = (1 << lvl) - 1 + (i & ((1 << lvl) - 1));
                float d = decw[r * DSTRIDE + node];
                p *= ((i >> lvl) & 1) ? (1.0f - d) : d;
            }
            float d5 = decw[r * DSTRIDE + 31 + i];
            routw[r * DSTRIDE + i] = p * d5;
            routw[r * DSTRIDE + i + 32] = p * (1.0f - d5);
        }
        wavebar();
        wait_async0();   // leaf tile must have landed before the f32 WMMAs

        // ---- leaf reduction: routing[16x64] @ lv[64x16] via f32 WMMA
        v8f po = (v8f){};
#pragma unroll
        for (int ks = 0; ks < 16; ++ks) {
            int k0 = ks * 4 + half * 2;
            v2f A, Bv;
            A.x = routw[l16 * DSTRIDE + k0];
            A.y = routw[l16 * DSTRIDE + k0 + 1];
            Bv.x = lvw[k0 * 16 + l16];
            Bv.y = lvw[(k0 + 1) * 16 + l16];
            po = wmma_f32x4(A, Bv, po);
        }

        // ---- attn-weighted accumulate
#pragma unroll
        for (int r = 0; r < 8; ++r) {
            int row16 = r + 8 * half;
            float w = attn_s[row16 * 64 + t] * coef;
            atomicAdd(&pred_s[row16 * 16 + l16], po[r] * w);
        }
        wavebar();
    }
    __syncthreads();

    // ---- final per-row softmax over 10 classes
    if (tid < 16) {
        float v[NCLS], mx = -1e30f;
#pragma unroll
        for (int c = 0; c < NCLS; ++c) {
            v[c] = pred_s[tid * 16 + c];
            mx = fmaxf(mx, v[c]);
        }
        float s = 0.0f;
#pragma unroll
        for (int c = 0; c < NCLS; ++c) {
            v[c] = __expf(v[c] - mx);
            s += v[c];
        }
        float inv = 1.0f / s;
#pragma unroll
        for (int c = 0; c < NCLS; ++c)
            out[(size_t)(row0 + tid) * NCLS + c] = v[c] * inv;
    }
}

// ---------------------------------------------------------------------------
extern "C" void kernel_launch(void* const* d_in, const int* in_sizes, int n_in,
                              void* d_out, int out_size, void* d_ws, size_t ws_size,
                              hipStream_t stream) {
    const float* x     = (const float*)d_in[0];
    const float* w1    = (const float*)d_in[1];
    const float* b1    = (const float*)d_in[2];
    const float* gamma = (const float*)d_in[3];
    const float* beta  = (const float*)d_in[4];
    const float* mean  = (const float*)d_in[5];
    const float* var   = (const float*)d_in[6];
    const float* w2    = (const float*)d_in[7];
    const float* b2    = (const float*)d_in[8];
    const float* tw    = (const float*)d_in[9];
    const float* tb    = (const float*)d_in[10];
    const float* temp  = (const float*)d_in[11];
    const float* lv    = (const float*)d_in[12];
    const float* rw    = (const float*)d_in[13];
    float* out = (float*)d_out;

    char* ws = (char*)d_ws;
    size_t off = 0;
    auto alloc = [&](size_t bytes) -> void* {
        void* p = ws + off;
        off = (off + bytes + 255) & ~(size_t)255;
        return p;
    };
    bf16_t* xhi   = (bf16_t*)alloc((size_t)NBATCH * DIN * 2);
    bf16_t* xlo   = (bf16_t*)alloc((size_t)NBATCH * DIN * 2);
    bf16_t* twhi  = (bf16_t*)alloc((size_t)NTREES * 64 * DIN * 2);
    bf16_t* twlo  = (bf16_t*)alloc((size_t)NTREES * 64 * DIN * 2);
    bf16_t* w1thi = (bf16_t*)alloc((size_t)NHID * DIN * 2);
    bf16_t* w1tlo = (bf16_t*)alloc((size_t)NHID * DIN * 2);
    bf16_t* w2thi = (bf16_t*)alloc((size_t)NTREES * NHID * 2);
    bf16_t* w2tlo = (bf16_t*)alloc((size_t)NTREES * NHID * 2);
    float*  bns   = (float*)alloc(NHID * 4);
    float*  bnt   = (float*)alloc(NHID * 4);
    unsigned* missing = (unsigned*)alloc((size_t)NBATCH * 4);
    float*  attnbuf = (float*)alloc((size_t)NBATCH * NTREES * 4);
    float*  lvpad   = (float*)alloc((size_t)NTREES * NLEAF * 16 * 4);
    (void)ws_size; (void)in_sizes; (void)n_in; (void)out_size;

    pack_x_kernel<<<NBATCH / 4, 128, 0, stream>>>(x, xhi, xlo, missing);
    pack_tw_kernel<<<(NTREES * 64 * DIN) / 128, 128, 0, stream>>>(tw, twhi, twlo);
    pack_lv_kernel<<<(NTREES * NLEAF * 16) / 128, 128, 0, stream>>>(lv, lvpad);
    pack_small_kernel<<<(DIN * NHID + NTREES * NHID + NHID + 127) / 128, 128, 0,
                        stream>>>(w1, w2, gamma, beta, mean, var, w1thi, w1tlo,
                                  w2thi, w2tlo, bns, bnt);
    attn_kernel<<<NBATCH / 64, 128, 0, stream>>>(xhi, xlo, w1thi, w1tlo, w2thi,
                                                 w2tlo, b1, bns, bnt, b2, attnbuf);
    tree_main_kernel<<<NBATCH / 16, 128, 0, stream>>>(
        xhi, xlo, twhi, twlo, attnbuf, missing, tb, temp, rw, lvpad, out);
}
